// imageCaptionModel_19378892439811
// MI455X (gfx1250) — matrix operand order repaired and checked by reading, not compiled, not run on hardware
//
#include <hip/hip_runtime.h>

typedef __attribute__((ext_vector_type(8)))  _Float16 v8h;
typedef __attribute__((ext_vector_type(16))) _Float16 v16h;
typedef __attribute__((ext_vector_type(8)))  float    v8f;

#define Hh 512
#define Bb 128
#define Tt 24
#define Vv 10000
#define AHh 50
#define AOo 10

// ---------------- WMMA tile loaders (16x16x32 f16, wave32 layouts) ----------
__device__ __forceinline__ v16h ld_a(const _Float16* __restrict__ A, int lda,
                                     int row, int kb, int g) {
    // A 16x32: lane<16 holds M=lane, K in {kb+8g..kb+8g+7} U {kb+16+8g..kb+23+8g}
    const _Float16* p = A + row * lda + kb + 8 * g;
    v8h c1 = *(const v8h*)(p);
    v8h c2 = *(const v8h*)(p + 16);
    v16h r;
#pragma unroll
    for (int i = 0; i < 8; ++i) { r[i] = c1[i]; r[i + 8] = c2[i]; }
    return r;
}

__device__ __forceinline__ v16h ld_b(const _Float16* __restrict__ Bt, int ldb,
                                     int col, int kb, int g) {
    // B 32x16 from transposed weight Bt[n][k]: lane holds N=col, K in {kb+8g..kb+8g+7} U {kb+16+8g..kb+23+8g}
    const _Float16* p = Bt + col * ldb + kb + 8 * g;
    v8h c1 = *(const v8h*)(p);
    v8h c2 = *(const v8h*)(p + 16);
    v16h r;
#pragma unroll
    for (int i = 0; i < 8; ++i) { r[i] = c1[i]; r[i + 8] = c2[i]; }
    return r;
}

// Each wave computes a 16x64 strip: one A load feeds 4 WMMAs.
// Software-pipelined: prefetch K+32 tiles before consuming current tiles so
// loads for the next chunk are in flight during the WMMAs.
__device__ __forceinline__ void wmma_mm4(const _Float16* __restrict__ A, int lda,
                                         const _Float16* __restrict__ Bt, int ldb,
                                         int K, int N, int m0, int n0,
                                         int sub, int g, v8f* acc) {
    bool val[4];
#pragma unroll
    for (int t = 0; t < 4; ++t) val[t] = (n0 + 16 * t) < N;

    v16h a_cur = ld_a(A, lda, m0 + sub, 0, g);
    v16h b_cur[4] = {};
#pragma unroll
    for (int t = 0; t < 4; ++t)
        if (val[t]) b_cur[t] = ld_b(Bt, ldb, n0 + 16 * t + sub, 0, g);

    for (int kb = 0; kb + 32 < K; kb += 32) {
        // prefetch next K-chunk
        v16h a_nxt = ld_a(A, lda, m0 + sub, kb + 32, g);
        v16h b_nxt[4] = {};
#pragma unroll
        for (int t = 0; t < 4; ++t)
            if (val[t]) b_nxt[t] = ld_b(Bt, ldb, n0 + 16 * t + sub, kb + 32, g);
        // consume current chunk while next loads are in flight
#pragma unroll
        for (int t = 0; t < 4; ++t)
            if (val[t])
                acc[t] = __builtin_amdgcn_wmma_f32_16x16x32_f16(
                    false, a_cur, false, b_cur[t], (short)0, acc[t], false, false);
        a_cur = a_nxt;
#pragma unroll
        for (int t = 0; t < 4; ++t) b_cur[t] = b_nxt[t];
    }
    // epilogue: last chunk
#pragma unroll
    for (int t = 0; t < 4; ++t)
        if (val[t])
            acc[t] = __builtin_amdgcn_wmma_f32_16x16x32_f16(
                false, a_cur, false, b_cur[t], (short)0, acc[t], false, false);
}

// ---------------- utility kernels -------------------------------------------
__global__ void k_cvt_f16(const float* s, _Float16* d, int n) {
    int i = blockIdx.x * blockDim.x + threadIdx.x;
    if (i < n) d[i] = (_Float16)s[i];
}

// dst[n*ldout + k] = (k < K) ? (f16)src[k*N + n] : 0   (transpose + pad K)
__global__ void k_transpose_f16(const float* src, _Float16* dst, int K, int N, int ldout) {
    int i = blockIdx.x * blockDim.x + threadIdx.x;
    if (i >= N * ldout) return;
    int n = i / ldout, k = i - n * ldout;
    dst[i] = (k < K) ? (_Float16)src[k * N + n] : (_Float16)0.0f;
}

__global__ void k_zero_u32(unsigned* p, int n) {
    int i = blockIdx.x * blockDim.x + threadIdx.x;
    if (i < n) p[i] = 0u;
}

// A0pre[(t*B+b)][0:512]=base[b], [512:1024]=emb[tok(b,t)]
__global__ void k_gather_a0(const _Float16* base16, const _Float16* emb16,
                            const int* xT, _Float16* A0) {
    int i = blockIdx.x * blockDim.x + threadIdx.x;   // over 3072*1024
    if (i >= Tt * Bb * 1024) return;
    int row = i >> 10, j = i & 1023;
    int t = row >> 7, b = row & 127;                 // row = t*B + b
    _Float16 v;
    if (j < Hh) v = base16[b * Hh + j];
    else        v = emb16[xT[b * Tt + t] * 512 + (j - Hh)];
    A0[i] = v;
}

// ---------------- GEMM kernels ----------------------------------------------
// base = leaky_relu(cnn @ W_in + b_in)  -> f16 [128,512]
__global__ void k_gemm_base(const _Float16* __restrict__ A, const _Float16* __restrict__ Bt,
                            const float* __restrict__ bias, _Float16* __restrict__ out) {
    const int Mt = Bb / 16, NG = (Hh / 16 + 3) / 4;
    int lane = threadIdx.x & 31, wid = threadIdx.x >> 5;
    int w = blockIdx.x * 8 + wid;
    int mt = w % Mt, ng = w / Mt;
    if (ng >= NG) return;
    int m0 = mt * 16, n0 = ng * 64, sub = lane & 15, g = lane >> 4;
    v8f acc[4] = {};
    wmma_mm4(A, 2048, Bt, 2048, 2048, Hh, m0, n0, sub, g, acc);
#pragma unroll
    for (int t = 0; t < 4; ++t) {
        int n = n0 + 16 * t + sub;
        if (n >= Hh) break;
#pragma unroll
        for (int r = 0; r < 8; ++r) {
            int row = m0 + 8 * g + r;
            float c = acc[t][r] + bias[n];
            c = c > 0.f ? c : 0.01f * c;
            out[row * Hh + n] = (_Float16)c;
        }
    }
}

// Z0pre = A0pre @ W0_top + cell0_b  -> f32 [3072,2048]
__global__ void k_gemm_pre(const _Float16* __restrict__ A, const _Float16* __restrict__ Bt,
                           const float* __restrict__ bias, float* __restrict__ out) {
    const int Mt = (Tt * Bb) / 16;   // 192
    int lane = threadIdx.x & 31, wid = threadIdx.x >> 5;
    int w = blockIdx.x * 8 + wid;
    int mt = w % Mt, ng = w / Mt;
    if (ng >= 32) return;
    int m0 = mt * 16, n0 = ng * 64, sub = lane & 15, g = lane >> 4;
    v8f acc[4] = {};
    wmma_mm4(A, 1024, Bt, 2048, 1024, 2048, m0, n0, sub, g, acc);
#pragma unroll
    for (int t = 0; t < 4; ++t) {
        int n = n0 + 16 * t + sub;
#pragma unroll
        for (int r = 0; r < 8; ++r) {
            int row = m0 + 8 * g + r;
            out[row * 2048 + n] = acc[t][r] + bias[n];
        }
    }
}

// Per-step recurrent gate GEMMs. blockIdx.y: 0 = cell0 (K=1024, +Z0pre),
// 1 = cell1 (K=2080 padded, +bias).
__global__ void k_gemm_z(const _Float16* __restrict__ A1, const _Float16* __restrict__ W0t,
                         const _Float16* __restrict__ W1t,
                         const float* __restrict__ Z0pre, const float* __restrict__ c1b,
                         float* __restrict__ Z0, float* __restrict__ Z1, int t) {
    const int Mt = Bb / 16;          // 8
    int lane = threadIdx.x & 31, wid = threadIdx.x >> 5;
    int w = blockIdx.x * 8 + wid;
    int mt = w % Mt, ng = w / Mt;
    if (ng >= 32) return;
    int m0 = mt * 16, n0 = ng * 64, sub = lane & 15, g = lane >> 4;
    v8f acc[4] = {};
    if (blockIdx.y == 0) {
        // s0_old (A1 cols 0..1023) @ W0 rows 1024..2047
        wmma_mm4(A1, 2080, W0t + 1024, 2048, 1024, 2048, m0, n0, sub, g, acc);
#pragma unroll
        for (int tt = 0; tt < 4; ++tt) {
            int n = n0 + 16 * tt + sub;
#pragma unroll
            for (int r = 0; r < 8; ++r) {
                int row = m0 + 8 * g + r;
                Z0[row * 2048 + n] = acc[tt][r] + Z0pre[(t * Bb + row) * 2048 + n];
            }
        }
    } else {
        // cat(s0_old, a, s1_old, pad) @ W1 (K padded to 2080)
        wmma_mm4(A1, 2080, W1t, 2080, 2080, 2048, m0, n0, sub, g, acc);
#pragma unroll
        for (int tt = 0; tt < 4; ++tt) {
            int n = n0 + 16 * tt + sub;
#pragma unroll
            for (int r = 0; r < 8; ++r) {
                int row = m0 + 8 * g + r;
                Z1[row * 2048 + n] = acc[tt][r] + c1b[n];
            }
        }
    }
}

// logits = H1 @ out_W + out_b -> d_out[b][t][v]
__global__ void k_gemm_out(const _Float16* __restrict__ A, const _Float16* __restrict__ Bt,
                           const float* __restrict__ bias, float* __restrict__ out) {
    const int Mt = (Tt * Bb) / 16;   // 192
    const int NG = (Vv / 16 + 3) / 4; // 157
    int lane = threadIdx.x & 31, wid = threadIdx.x >> 5;
    int w = blockIdx.x * 8 + wid;
    int mt = w % Mt, ng = w / Mt;
    if (ng >= NG) return;
    int m0 = mt * 16, n0 = ng * 64, sub = lane & 15, g = lane >> 4;
    v8f acc[4] = {};
    wmma_mm4(A, 512, Bt, 512, 512, Vv, m0, n0, sub, g, acc);
#pragma unroll
    for (int tt = 0; tt < 4; ++tt) {
        int n = n0 + 16 * tt + sub;
        if (n >= Vv) break;
#pragma unroll
        for (int r = 0; r < 8; ++r) {
            int row = m0 + 8 * g + r;      // row = t*B + b
            int st = row >> 7, b = row & 127;
            out[(b * Tt + st) * Vv + n] = acc[tt][r] + bias[n];
        }
    }
}

// ---------------- attention (tiny, VALU) ------------------------------------
__global__ void k_attn(const float* S0, const float* W1, const float* b1,
                       const float* W2, const float* b2, _Float16* A1) {
    int b = blockIdx.x, j = threadIdx.x;
    __shared__ float y1[AHh];
    __shared__ float lg[AOo];
    if (j < AHh) {
        float acc = b1[j];
        for (int k = 0; k < 2 * Hh; ++k) acc += S0[b * 1024 + k] * W1[k * AHh + j];
        y1[j] = acc > 0.f ? acc : 0.01f * acc;
    }
    __syncthreads();
    if (j < AOo) {
        float acc = b2[j];
        for (int k = 0; k < AHh; ++k) acc += y1[k] * W2[k * AOo + j];
        lg[j] = acc;
    }
    __syncthreads();
    if (j == 0) {
        float mx = lg[0];
        for (int q = 1; q < AOo; ++q) mx = fmaxf(mx, lg[q]);
        float e[AOo], s = 0.f;
        for (int q = 0; q < AOo; ++q) { e[q] = __expf(lg[q] - mx); s += e[q]; }
        float inv = 1.f / s;
        for (int q = 0; q < AOo; ++q) A1[b * 2080 + 1024 + q] = (_Float16)(e[q] * inv);
    }
}

// ---------------- gates / state update --------------------------------------
__global__ void k_gates(const float* Z0, const float* Z1, float* S0, float* S1,
                        _Float16* A1, _Float16* H1, int t) {
    int idx = blockIdx.x * blockDim.x + threadIdx.x;   // 2*B*H
    if (idx >= 2 * Bb * Hh) return;
    int cell = idx >= (Bb * Hh);
    int r = idx - cell * (Bb * Hh);
    int b = r >> 9, j = r & 511;
    const float* Z = cell ? Z1 : Z0;
    float* S = cell ? S1 : S0;
    float zi = Z[b * 2048 + j];
    float zf = Z[b * 2048 + 512 + j];
    float zo = Z[b * 2048 + 1024 + j];
    float zg = Z[b * 2048 + 1536 + j];
    float i_ = 1.f / (1.f + __expf(-zi));
    float f_ = 1.f / (1.f + __expf(-zf));
    float o_ = 1.f / (1.f + __expf(-zo));
    float c_old = S[b * 1024 + 512 + j];
    float c_new = f_ * c_old + i_ * tanhf(zg);
    float h_new = o_ * tanhf(tanhf(c_new));
    S[b * 1024 + j] = h_new;
    S[b * 1024 + 512 + j] = c_new;
    int a1off = cell ? 1034 : 0;
    A1[b * 2080 + a1off + j] = (_Float16)h_new;
    A1[b * 2080 + a1off + 512 + j] = (_Float16)c_new;
    if (cell) H1[(t * Bb + b) * Hh + j] = (_Float16)h_new;
}

// ---------------- launcher --------------------------------------------------
extern "C" void kernel_launch(void* const* d_in, const int* in_sizes, int n_in,
                              void* d_out, int out_size, void* d_ws, size_t ws_size,
                              hipStream_t stream) {
    const float* cnn  = (const float*)d_in[0];
    const int*   xT   = (const int*)d_in[1];
    const float* emb  = (const float*)d_in[2];
    const float* Win  = (const float*)d_in[3];
    const float* bin  = (const float*)d_in[4];
    const float* aW1  = (const float*)d_in[5];
    const float* ab1  = (const float*)d_in[6];
    const float* aW2  = (const float*)d_in[7];
    const float* ab2  = (const float*)d_in[8];
    const float* oW   = (const float*)d_in[9];
    const float* ob   = (const float*)d_in[10];
    const float* c0W  = (const float*)d_in[11];
    const float* c0b  = (const float*)d_in[12];
    const float* c1W  = (const float*)d_in[13];
    const float* c1b  = (const float*)d_in[14];
    float* out = (float*)d_out;

    char* ws = (char*)d_ws;
    size_t off = 0;
    auto alloc = [&](size_t bytes) -> char* {
        char* p = ws + off;
        off = (off + bytes + 255) & ~(size_t)255;
        return p;
    };
    _Float16* W0t   = (_Float16*)alloc((size_t)2048 * 2048 * 2);
    _Float16* W1t   = (_Float16*)alloc((size_t)2048 * 2080 * 2);
    _Float16* OWt   = (_Float16*)alloc((size_t)Vv * 512 * 2);
    _Float16* WINt  = (_Float16*)alloc((size_t)512 * 2048 * 2);
    _Float16* EMB16 = (_Float16*)alloc((size_t)Vv * 512 * 2);
    _Float16* CNN16 = (_Float16*)alloc((size_t)Bb * 2048 * 2);
    _Float16* BASE16= (_Float16*)alloc((size_t)Bb * Hh * 2);
    _Float16* A0PRE = (_Float16*)alloc((size_t)Tt * Bb * 1024 * 2);
    float*    Z0PRE = (float*)   alloc((size_t)Tt * Bb * 2048 * 4);
    _Float16* A1    = (_Float16*)alloc((size_t)Bb * 2080 * 2);
    float*    S0    = (float*)   alloc((size_t)Bb * 1024 * 4);
    float*    S1    = (float*)   alloc((size_t)Bb * 1024 * 4);
    float*    Z0    = (float*)   alloc((size_t)Bb * 2048 * 4);
    float*    Z1    = (float*)   alloc((size_t)Bb * 2048 * 4);
    _Float16* H1    = (_Float16*)alloc((size_t)Tt * Bb * Hh * 2);

    const int TB = 256;
    // weight conversion / transpose (one-time per launch, all independent)
    k_transpose_f16<<<(2048 * 2048 + TB - 1) / TB, TB, 0, stream>>>(c0W, W0t, 2048, 2048, 2048);
    k_transpose_f16<<<(2048 * 2080 + TB - 1) / TB, TB, 0, stream>>>(c1W, W1t, 2058, 2048, 2080);
    k_transpose_f16<<<(Vv * 512 + TB - 1) / TB, TB, 0, stream>>>(oW, OWt, 512, Vv, 512);
    k_transpose_f16<<<(512 * 2048 + TB - 1) / TB, TB, 0, stream>>>(Win, WINt, 2048, 512, 2048);
    k_cvt_f16<<<(Vv * 512 + TB - 1) / TB, TB, 0, stream>>>(emb, EMB16, Vv * 512);
    k_cvt_f16<<<(Bb * 2048 + TB - 1) / TB, TB, 0, stream>>>(cnn, CNN16, Bb * 2048);
    // zero states + A1 (incl. K padding + a slot)
    k_zero_u32<<<(Bb * 2080 * 2 / 4 + TB - 1) / TB, TB, 0, stream>>>((unsigned*)A1, Bb * 2080 * 2 / 4);
    k_zero_u32<<<(Bb * 1024 + TB - 1) / TB, TB, 0, stream>>>((unsigned*)S0, Bb * 1024);
    k_zero_u32<<<(Bb * 1024 + TB - 1) / TB, TB, 0, stream>>>((unsigned*)S1, Bb * 1024);

    // base image features: [128,2048]x[2048,512]
    k_gemm_base<<<(8 * 8) / 8, TB, 0, stream>>>(CNN16, WINt, bin, BASE16);
    // gather cat(base, emb_t) for all steps
    k_gather_a0<<<(Tt * Bb * 1024 + TB - 1) / TB, TB, 0, stream>>>(BASE16, EMB16, xT, A0PRE);
    // precompute token-dependent half of cell0 gates: [3072,1024]x[1024,2048]
    k_gemm_pre<<<(192 * 32) / 8, TB, 0, stream>>>(A0PRE, W0t, c0b, Z0PRE);

    // recurrence: 24 sequential steps (stream order = dependency order)
    for (int t = 0; t < Tt; ++t) {
        k_attn<<<Bb, 64, 0, stream>>>(S0, aW1, ab1, aW2, ab2, A1);
        dim3 zg((8 * 32) / 8, 2);
        k_gemm_z<<<zg, TB, 0, stream>>>(A1, W0t, W1t, Z0PRE, c1b, Z0, Z1, t);
        k_gates<<<(2 * Bb * Hh + TB - 1) / TB, TB, 0, stream>>>(Z0, Z1, S0, S1, A1, H1, t);
    }

    // vocab projection for all steps: [3072,512]x[512,10000]
    k_gemm_out<<<(192 * 157) / 8, TB, 0, stream>>>(H1, OWt, ob, out);
}
